// MultiHeadLatentAttention_41283225649393
// MI455X (gfx1250) — matrix operand order repaired and checked by reading, not compiled, not run on hardware
//
#include <hip/hip_runtime.h>
#include <hip/hip_bf16.h>
#include <math.h>
#include <stdint.h>

// ---------------------------------------------------------------------------
// MLA forward for MI455X (gfx1250, wave32, WMMA).
// All matmuls via v_wmma_f32_16x16x32_bf16. Flash-attention keeps the
// SEQ x SEQ score matrix in registers (never hits HBM).  K tiles are staged
// into LDS with the CDNA5 async-DMA path (global_load_async_to_lds_b128,
// ASYNCcnt), V tiles go through VGPRs because they are transposed on the fly.
// ---------------------------------------------------------------------------

#define SEQ   4096
#define EMBED 2048
#define NH    16
#define QLR   512
#define KVLR  512
#define RD    64
#define VD    128
#define KD    192   // VD + RD

typedef __bf16 bf16_t;
typedef __attribute__((ext_vector_type(16))) __bf16 v16bf;
typedef __attribute__((ext_vector_type(8)))  __bf16 v8bf;
typedef __attribute__((ext_vector_type(8)))  float  v8f;

// ---- WMMA fragment loaders (ISA 7.12.2 layouts, wave32) -------------------
// A (16x32 bf16): lane L holds row M=L&15; K chunks [hi*8..hi*8+7] and
// [16+hi*8..16+hi*8+7] where hi=L>>4.  Two 16B loads per lane.
static __device__ inline v16bf load_a_frag(const bf16_t* base, int strideElems) {
  const int lane = threadIdx.x & 31;
  const bf16_t* p = base + (size_t)(lane & 15) * strideElems + ((lane >> 4) << 3);
  union { v16bf v; v8bf h[2]; } u;
  u.h[0] = *reinterpret_cast<const v8bf*>(p);
  u.h[1] = *reinterpret_cast<const v8bf*>(p + 16);
  return u.v;
}

// B (32x16 bf16, column n contiguous along K in memory): lane L holds column
// N=L&15; K chunks [hi*8..hi*8+7] and [16+hi*8..16+hi*8+7].  Two 16B loads.
static __device__ inline v16bf load_b_frag(const bf16_t* base, int strideElems) {
  const int lane = threadIdx.x & 31;
  const bf16_t* p = base + (size_t)(lane & 15) * strideElems + ((lane >> 4) << 3);
  union { v16bf v; v8bf h[2]; } u;
  u.h[0] = *reinterpret_cast<const v8bf*>(p);
  u.h[1] = *reinterpret_cast<const v8bf*>(p + 16);
  return u.v;
}

static __device__ inline v8f wmma_bf16(v16bf a, v16bf b, v8f c) {
  // (neg_a, A, neg_b, B, c_mod, C, reuse_a, reuse_b)
  return __builtin_amdgcn_wmma_f32_16x16x32_bf16(false, a, false, b, (short)0, c,
                                                 false, false);
}

// CDNA5 async copy: global -> LDS, 16B per lane, tracked by ASYNCcnt.
static __device__ inline void async_copy_b128(unsigned ldsByteAddr,
                                              unsigned gvByteOff,
                                              unsigned long long gbase) {
  asm volatile("global_load_async_to_lds_b128 %0, %1, %2"
               :: "v"(ldsByteAddr), "v"(gvByteOff), "s"(gbase)
               : "memory");
}
static __device__ inline void wait_asynccnt0() {
  asm volatile("s_wait_asynccnt 0x0" ::: "memory");
}

// ---- fp32 -> bf16 convert --------------------------------------------------
__global__ void k_f32_to_bf16(const float* __restrict__ in, bf16_t* __restrict__ out,
                              int n) {
  int stride = gridDim.x * blockDim.x;
  for (int i = blockIdx.x * blockDim.x + threadIdx.x; i < n; i += stride)
    out[i] = (bf16_t)in[i];
}

// ---- Generic WMMA GEMM: C[M,N] = A[M,K] * B[N,K]^T -------------------------
// Block = 256 threads = 8 waves; block tile 128(M) x 64(N); wave tile 32x32.
enum { OUT_F32 = 0, OUT_BF16 = 1, OUT_QSPLIT = 2, OUT_KVSPLIT = 3 };

template <int MODE>
__global__ __launch_bounds__(256) void k_gemm(const bf16_t* __restrict__ A,
                                              const bf16_t* __restrict__ B,
                                              float* __restrict__ outF,
                                              bf16_t* __restrict__ outB,
                                              bf16_t* __restrict__ outB2,
                                              int M, int N, int K) {
  const int wave  = threadIdx.x >> 5;
  const int lane  = threadIdx.x & 31;
  const int waveM = wave >> 1;          // 0..3
  const int waveN = wave & 1;           // 0..1
  const int m0 = blockIdx.y * 128 + waveM * 32;
  const int n0 = blockIdx.x * 64  + waveN * 32;

  v8f acc[2][2] = {};
  const bf16_t* aBase = A + (size_t)m0 * K;
  const bf16_t* bBase = B + (size_t)n0 * K;

  for (int k = 0; k < K; k += 32) {
    if (k + 256 < K) {                      // global_prefetch_b8 hint
      __builtin_prefetch(aBase + k + 256);
      __builtin_prefetch(bBase + k + 256);
    }
    v16bf a0 = load_a_frag(aBase + k, K);
    v16bf a1 = load_a_frag(aBase + (size_t)16 * K + k, K);
    v16bf b0 = load_b_frag(bBase + k, K);
    v16bf b1 = load_b_frag(bBase + (size_t)16 * K + k, K);
    acc[0][0] = wmma_bf16(a0, b0, acc[0][0]);
    acc[0][1] = wmma_bf16(a0, b1, acc[0][1]);
    acc[1][0] = wmma_bf16(a1, b0, acc[1][0]);
    acc[1][1] = wmma_bf16(a1, b1, acc[1][1]);
  }

  const int hi = lane >> 4, nl = lane & 15;
#pragma unroll
  for (int i = 0; i < 2; ++i)
#pragma unroll
    for (int j = 0; j < 2; ++j)
#pragma unroll
      for (int r = 0; r < 8; ++r) {
        const int m = m0 + i * 16 + hi * 8 + r;
        const int n = n0 + j * 16 + nl;
        const float v = acc[i][j][r];
        if (MODE == OUT_F32) {
          outF[(size_t)m * N + n] = v;
        } else if (MODE == OUT_BF16) {
          outB[(size_t)m * N + n] = (bf16_t)v;
        } else if (MODE == OUT_QSPLIT) {
          // scatter into Qf[h][m][d], d in [0,128)
          const int h = n >> 7, d = n & 127;
          outB[((size_t)h * SEQ + m) * KD + d] = (bf16_t)v;
        } else {  // OUT_KVSPLIT: first half -> Kf content, second -> V
          const int half = N >> 1;
          if (n < half) {
            const int h = n >> 7, d = n & 127;
            outB[((size_t)h * SEQ + m) * KD + d] = (bf16_t)v;
          } else {
            const int n2 = n - half;
            const int h = n2 >> 7, d = n2 & 127;
            outB2[((size_t)h * SEQ + m) * VD + d] = (bf16_t)v;
          }
        }
      }
}

// ---- RoPE + assemble rope columns of Qf / Kf -------------------------------
__global__ __launch_bounds__(256) void k_rope_assemble(const bf16_t* __restrict__ qr,
                                                       const bf16_t* __restrict__ kr,
                                                       bf16_t* __restrict__ Qf,
                                                       bf16_t* __restrict__ Kf) {
  const int s = blockIdx.x;
  const float fs = (float)s;
  // q_rope: per (head, j) pair, j in [0,32)
  for (int i = threadIdx.x; i < NH * 32; i += blockDim.x) {
    const int h = i >> 5, j = i & 31;
    const float invf = __expf(-(float)j * (9.210340371976184f / 32.0f)); // 10000^(-j/32)
    float sn, cs;
    __sincosf(fs * invf, &sn, &cs);
    const float x1 = (float)qr[(size_t)s * (NH * RD) + h * RD + j];
    const float x2 = (float)qr[(size_t)s * (NH * RD) + h * RD + j + 32];
    Qf[((size_t)h * SEQ + s) * KD + VD + j]      = (bf16_t)(x1 * cs - x2 * sn);
    Qf[((size_t)h * SEQ + s) * KD + VD + 32 + j] = (bf16_t)(x2 * cs + x1 * sn);
  }
  // k_rope: shared across heads
  if (threadIdx.x < 32) {
    const int j = threadIdx.x;
    const float invf = __expf(-(float)j * (9.210340371976184f / 32.0f));
    float sn, cs;
    __sincosf(fs * invf, &sn, &cs);
    const float x1 = (float)kr[(size_t)s * RD + j];
    const float x2 = (float)kr[(size_t)s * RD + j + 32];
    const bf16_t o1 = (bf16_t)(x1 * cs - x2 * sn);
    const bf16_t o2 = (bf16_t)(x2 * cs + x1 * sn);
#pragma unroll
    for (int h = 0; h < NH; ++h) {
      Kf[((size_t)h * SEQ + s) * KD + VD + j]      = o1;
      Kf[((size_t)h * SEQ + s) * KD + VD + 32 + j] = o2;
    }
  }
}

// ---- Flash attention (causal) ---------------------------------------------
// grid = (SEQ/128, NH); block = 256 (8 waves, 16 query rows each).
// Key tiles of 32 staged in LDS (K via async DMA); scores & O stay in
// WMMA accumulators.
#define QT 128
#define KT 32

__global__ __launch_bounds__(256) void k_flash(const bf16_t* __restrict__ Qf,
                                               const bf16_t* __restrict__ Kf,
                                               const bf16_t* __restrict__ Vv,
                                               bf16_t* __restrict__ attn) {
  __shared__ bf16_t sK[KT * KD];      // 32 keys x 192, row-major (12 KB)
  __shared__ bf16_t sVT[VD * KT];     // V transposed: [vd][k] (8 KB)
  __shared__ bf16_t sP[8][16 * KT];   // per-wave P staging (8 KB)

  const int wave  = threadIdx.x >> 5;
  const int lane  = threadIdx.x & 31;
  const int head  = blockIdx.y;
  const int qBase = blockIdx.x * QT;
  const int qWave = qBase + wave * 16;
  const int hi = lane >> 4, nl = lane & 15;

  // Q fragments for this wave's 16 rows: 6 chunks of K=32 over KD=192.
  const bf16_t* qPtr = Qf + ((size_t)head * SEQ + qWave) * KD;
  v16bf qf[6];
#pragma unroll
  for (int d = 0; d < 6; ++d) qf[d] = load_a_frag(qPtr + d * 32, KD);

  v8f accO[8] = {};
  float mRun[8], lRun[8];
#pragma unroll
  for (int r = 0; r < 8; ++r) { mRun[r] = -1e30f; lRun[r] = 0.f; }

  const bf16_t* kHead = Kf + (size_t)head * SEQ * KD;
  const bf16_t* vHead = Vv + (size_t)head * SEQ * VD;
  const unsigned sKlds = (unsigned)(uintptr_t)&sK[0];   // LDS byte offset
  const int nTiles = (qBase + QT) / KT;            // causal: keys <= block max q
  const float scale = 0.07216878364870323f;        // 1/sqrt(192)

  for (int kt = 0; kt < nTiles; ++kt) {
    const int kBase = kt * KT;
    __syncthreads();  // previous iteration's LDS readers are done
    {   // K tile (32 x 192 bf16 = 12 KB, contiguous rows): async DMA to LDS.
      const unsigned long long gbase =
          (unsigned long long)(uintptr_t)(kHead + (size_t)kBase * KD);
#pragma unroll
      for (int c = 0; c < 3; ++c) {
        const unsigned byteOff = ((unsigned)threadIdx.x * 3u + (unsigned)c) * 16u;
        async_copy_b128(sKlds + byteOff, byteOff, gbase);
      }
    }
    {   // V tile transposed into sVT[vd][k] (needs shuffle -> VGPR path)
      const int k  = threadIdx.x >> 3;
      const int vb = (threadIdx.x & 7) * 16;
      const bf16_t* src = vHead + (size_t)(kBase + k) * VD + vb;
      __align__(16) bf16_t tmp[16];
      *reinterpret_cast<v8bf*>(tmp)     = *reinterpret_cast<const v8bf*>(src);
      *reinterpret_cast<v8bf*>(tmp + 8) = *reinterpret_cast<const v8bf*>(src + 8);
#pragma unroll
      for (int j = 0; j < 16; ++j) sVT[(vb + j) * KT + k] = tmp[j];
    }
    wait_asynccnt0();   // this wave's async K-tile stores are in LDS
    __syncthreads();    // all waves' tiles visible

    // S = Q * K^T : two 16x16 score blocks, K-dim 192 = 6x32
    v8f s0 = {}, s1 = {};
#pragma unroll
    for (int d = 0; d < 6; ++d) {
      v16bf b0 = load_b_frag(&sK[0]       + d * 32, KD);
      v16bf b1 = load_b_frag(&sK[16 * KD] + d * 32, KD);
      s0 = wmma_bf16(qf[d], b0, s0);
      s1 = wmma_bf16(qf[d], b1, s1);
    }

    // Online softmax (per-row across lanes 0..15 of each half)
#pragma unroll
    for (int r = 0; r < 8; ++r) {
      const int q  = qWave + hi * 8 + r;
      const int k0 = kBase + nl;
      const int k1 = kBase + 16 + nl;
      float v0 = (k0 <= q) ? s0[r] * scale : -1e30f;
      float v1 = (k1 <= q) ? s1[r] * scale : -1e30f;
      float t = fmaxf(v0, v1);
#pragma unroll
      for (int off = 1; off < 16; off <<= 1) t = fmaxf(t, __shfl_xor(t, off, 32));
      const float mNew  = fmaxf(mRun[r], t);
      const float alpha = __expf(mRun[r] - mNew);
      const float p0 = (k0 <= q) ? __expf(v0 - mNew) : 0.f;
      const float p1 = (k1 <= q) ? __expf(v1 - mNew) : 0.f;
      float rs = p0 + p1;
#pragma unroll
      for (int off = 1; off < 16; off <<= 1) rs += __shfl_xor(rs, off, 32);
      lRun[r] = lRun[r] * alpha + rs;
      mRun[r] = mNew;
#pragma unroll
      for (int n = 0; n < 8; ++n) accO[n][r] *= alpha;
      sP[wave][(hi * 8 + r) * KT + nl]      = (bf16_t)p0;
      sP[wave][(hi * 8 + r) * KT + 16 + nl] = (bf16_t)p1;
    }
    __syncthreads();  // P stores (cross-lane) visible before A-frag reload

    // O += P * V  (K-dim = 32 keys, 8 column blocks of 16 over VD=128)
    v16bf pa = load_a_frag(&sP[wave][0], KT);
#pragma unroll
    for (int n = 0; n < 8; ++n) {
      v16bf bv = load_b_frag(&sVT[(n * 16) * KT], KT);
      accO[n] = wmma_bf16(pa, bv, accO[n]);
    }
  }

  // Epilogue: normalize, write bf16 into attn[q][h*128 + vd]
#pragma unroll
  for (int n = 0; n < 8; ++n)
#pragma unroll
    for (int r = 0; r < 8; ++r) {
      const int q = qWave + hi * 8 + r;
      const float v = accO[n][r] / lRun[r];
      attn[(size_t)q * (NH * VD) + head * VD + n * 16 + nl] = (bf16_t)v;
    }
}

// ---------------------------------------------------------------------------
extern "C" void kernel_launch(void* const* d_in, const int* in_sizes, int n_in,
                              void* d_out, int out_size, void* d_ws, size_t ws_size,
                              hipStream_t stream) {
  const float* x   = (const float*)d_in[0];
  // d_in[1] = causal mask (computed analytically, unused)
  const float* wqd = (const float*)d_in[2];   // (512, 2048)
  const float* wqu = (const float*)d_in[3];   // (2048, 512)
  const float* wqr = (const float*)d_in[4];   // (1024, 512)
  const float* wkd = (const float*)d_in[5];   // (512, 2048)
  const float* wku = (const float*)d_in[6];   // (4096, 512)
  const float* wkr = (const float*)d_in[7];   // (64, 2048)
  const float* wo  = (const float*)d_in[8];   // (2048, 2048)
  float* out = (float*)d_out;

  char* ws = (char*)d_ws;
  size_t off = 0;
  auto alloc = [&](size_t elems) -> bf16_t* {
    bf16_t* p = (bf16_t*)(ws + off);
    off += (elems * sizeof(bf16_t) + 255) & ~(size_t)255;
    return p;
  };
  bf16_t* xb    = alloc((size_t)SEQ * EMBED);
  bf16_t* wqd_b = alloc((size_t)QLR * EMBED);
  bf16_t* wqu_b = alloc((size_t)(NH * VD) * QLR);
  bf16_t* wqr_b = alloc((size_t)(NH * RD) * QLR);
  bf16_t* wkd_b = alloc((size_t)KVLR * EMBED);
  bf16_t* wku_b = alloc((size_t)(NH * 2 * VD) * KVLR);
  bf16_t* wkr_b = alloc((size_t)RD * EMBED);
  bf16_t* wo_b  = alloc((size_t)EMBED * (NH * VD));
  bf16_t* cq_b  = alloc((size_t)SEQ * QLR);
  bf16_t* ckv_b = alloc((size_t)SEQ * KVLR);
  bf16_t* qr_b  = alloc((size_t)SEQ * NH * RD);
  bf16_t* kr_b  = alloc((size_t)SEQ * RD);
  bf16_t* Qf    = alloc((size_t)NH * SEQ * KD);
  bf16_t* Kf    = alloc((size_t)NH * SEQ * KD);
  bf16_t* Vv    = alloc((size_t)NH * SEQ * VD);
  bf16_t* at_b  = alloc((size_t)SEQ * NH * VD);

  auto cvt = [&](const float* src, bf16_t* dst, size_t n) {
    int blocks = (int)((n + 1023) / 1024);
    if (blocks > 2048) blocks = 2048;
    k_f32_to_bf16<<<blocks, 256, 0, stream>>>(src, dst, (int)n);
  };
  cvt(x,   xb,    (size_t)SEQ * EMBED);
  cvt(wqd, wqd_b, (size_t)QLR * EMBED);
  cvt(wqu, wqu_b, (size_t)(NH * VD) * QLR);
  cvt(wqr, wqr_b, (size_t)(NH * RD) * QLR);
  cvt(wkd, wkd_b, (size_t)KVLR * EMBED);
  cvt(wku, wku_b, (size_t)(NH * 2 * VD) * KVLR);
  cvt(wkr, wkr_b, (size_t)RD * EMBED);
  cvt(wo,  wo_b,  (size_t)EMBED * (NH * VD));

  const dim3 blk(256);
  auto grid = [](int M, int N) { return dim3(N / 64, M / 128); };

  // c_q = x @ w_q_down^T ; c_kv = x @ w_kv_down^T
  k_gemm<OUT_BF16><<<grid(SEQ, QLR), blk, 0, stream>>>(xb, wqd_b, nullptr, cq_b,
                                                       nullptr, SEQ, QLR, EMBED);
  k_gemm<OUT_BF16><<<grid(SEQ, KVLR), blk, 0, stream>>>(xb, wkd_b, nullptr, ckv_b,
                                                        nullptr, SEQ, KVLR, EMBED);
  // q_content -> Qf[:, :, 0:128] (head-split epilogue)
  k_gemm<OUT_QSPLIT><<<grid(SEQ, NH * VD), blk, 0, stream>>>(cq_b, wqu_b, nullptr, Qf,
                                                             nullptr, SEQ, NH * VD, QLR);
  // q_rope (pre-rotation)
  k_gemm<OUT_BF16><<<grid(SEQ, NH * RD), blk, 0, stream>>>(cq_b, wqr_b, nullptr, qr_b,
                                                           nullptr, SEQ, NH * RD, QLR);
  // kv -> Kf content + V (split epilogue)
  k_gemm<OUT_KVSPLIT><<<grid(SEQ, NH * 2 * VD), blk, 0, stream>>>(
      ckv_b, wku_b, nullptr, Kf, Vv, SEQ, NH * 2 * VD, KVLR);
  // k_rope shared (pre-rotation)
  k_gemm<OUT_BF16><<<grid(SEQ, RD), blk, 0, stream>>>(xb, wkr_b, nullptr, kr_b,
                                                      nullptr, SEQ, RD, EMBED);
  // RoPE into Qf/Kf cols 128..191
  k_rope_assemble<<<SEQ, 256, 0, stream>>>(qr_b, kr_b, Qf, Kf);
  // Flash attention -> attn bf16 [SEQ][H*VD]
  k_flash<<<dim3(SEQ / QT, NH), 256, 0, stream>>>(Qf, Kf, Vv, at_b);
  // out = attn @ w_out^T (fp32 result)
  k_gemm<OUT_F32><<<grid(SEQ, EMBED), blk, 0, stream>>>(at_b, wo_b, out, nullptr,
                                                        nullptr, SEQ, EMBED, NH * VD);
  (void)in_sizes; (void)n_in; (void)out_size; (void)ws_size;
}
